// RNNModel_57131654971877
// MI455X (gfx1250) — compile-verified, matrix operand order checked by reading, not run on hardware
//
#include <hip/hip_runtime.h>

// 2-layer tanh RNN + linear head for MI455X (gfx1250, wave32, WMMA).
// Phase A: U0[t,b,:] = x[b,t,:] @ Wxh0^T + b0   (parallel WMMA GEMM, f16 out in d_ws)
// Phase B: fused sequential scan over t: 8 WGs (one per 16-row batch chunk),
//          all scan weights pinned in VGPRs (gfx1250 VGPR-MSB, no spills seen),
//          hidden state ping-ponged in LDS, U0 tiles DMA'd in with
//          global_load_async_to_lds_b128 one step ahead, 2 barriers per step.
// Needs T*B*H*2 = 32 MB of d_ws.

typedef __attribute__((ext_vector_type(16))) _Float16 v16h;
typedef __attribute__((ext_vector_type(4)))  _Float16 v4h;
typedef __attribute__((ext_vector_type(8)))  float    v8f;

#define T_STEPS 1024
#define BATCH   128
#define DH      256   // D == H == 256

__device__ __forceinline__ v8f wmma16(v16h a, v16h b, v8f c) {
  // D = A(16x32 f16) x B(32x16 f16) + C(16x16 f32)
  return __builtin_amdgcn_wmma_f32_16x16x32_f16(false, a, false, b, (short)0, c, false, false);
}

__device__ __forceinline__ v8f zero8() {
  v8f z = {0.f, 0.f, 0.f, 0.f, 0.f, 0.f, 0.f, 0.f};
  return z;
}

// Fast tanh: 1 - 2/(exp(2x)+1). v_exp_f32 / v_rcp_f32 are TRANS ops that
// co-execute with XDL WMMA; saturates correctly to +/-1 for large |x|.
__device__ __forceinline__ float fast_tanh(float x) {
  float e = __builtin_amdgcn_exp2f(x * 2.8853900817779268f);  // 2*log2(e)
  float r = __builtin_amdgcn_rcpf(e + 1.0f);
  return __builtin_fmaf(-2.0f, r, 1.0f);
}

// Async DMA: 16 bytes global -> LDS per lane (ASYNCcnt-tracked, no VGPR data).
__device__ __forceinline__ void async_copy16(void* lds_dst, const void* gsrc) {
  unsigned lo = (unsigned)(unsigned long long)lds_dst;  // LDS byte address
  asm volatile("global_load_async_to_lds_b128 %0, %1, off"
               :: "v"(lo), "v"(gsrc) : "memory");
}
__device__ __forceinline__ void wait_async0() {
  asm volatile("s_wait_asynccnt 0x0" ::: "memory");
}

// B-fragment for D = A x B with B = W^T (so B[k][n] = W[n][k]).
// B fragment, contiguous k as in load_afrag: lane l<16 holds col n = nbase+l, K = kc*32..+15;
// lanes 16..31 hold the same cols, K = kc*32+16..+31.
__device__ __forceinline__ v16h load_wfrag(const float* __restrict__ W, int nbase, int kc, int lane) {
  const int n = nbase + (lane & 15);
  const int k = kc * 32 + (lane >> 4) * 16;
  const float4* p = (const float4*)(W + (size_t)n * DH + k);
  v16h r;
  #pragma unroll
  for (int i = 0; i < 4; ++i) {
    float4 f = p[i];
    r[i*4+0] = (_Float16)f.x; r[i*4+1] = (_Float16)f.y;
    r[i*4+2] = (_Float16)f.z; r[i*4+3] = (_Float16)f.w;
  }
  return r;
}

// A-fragment from an LDS row-major [16][DH] f16 buffer.
// A fragment, contiguous k as in load_wfrag: lane l holds row m = l&15, 16 contiguous K at
// kc*32 + (l>=16 ? 16 : 0) -> one aligned 32-byte LDS read.
__device__ __forceinline__ v16h load_afrag(const _Float16* buf, int kc, int lane) {
  const int m = lane & 15;
  const int k = kc * 32 + (lane >> 4) * 16;
  return *(const v16h*)(buf + m * DH + k);
}

// ---------------------------------------------------------------------------
// Phase A: U0 = x @ Wxh0^T + b0, stored f16 as [T][B][H].
// Grid: (T, B/64); block: 256 threads (8 waves). Per WG: 64 rows x 256 cols.
// ---------------------------------------------------------------------------
__global__ __launch_bounds__(256, 1)
void u0_gemm_kernel(const float* __restrict__ x, const float* __restrict__ Wxh0,
                    const float* __restrict__ b0, _Float16* __restrict__ U0) {
  const int t   = blockIdx.x;     // timestep
  const int bb  = blockIdx.y;     // 64-row batch block
  const int tid = threadIdx.x, lane = tid & 31, wave = tid >> 5;
  const int nbase = wave * 32;    // this wave's 32 output columns

  __shared__ __attribute__((aligned(32))) _Float16 xt[64 * DH];

  // Stage A-tile rows b = bb*64..+63 of x[:, t, :] (f32 -> f16), coalesced.
  #pragma unroll
  for (int i = 0; i < 16; ++i) {
    int idx = i * 256 + tid;                 // float4 index within 64 x (DH/4)
    int row = idx >> 6;
    int c4  = idx & 63;
    float4 f = ((const float4*)(x + ((size_t)(bb*64 + row) * T_STEPS + t) * DH))[c4];
    v4h h; h[0]=(_Float16)f.x; h[1]=(_Float16)f.y; h[2]=(_Float16)f.z; h[3]=(_Float16)f.w;
    *(v4h*)&xt[row * DH + c4 * 4] = h;
  }

  // Weight fragments for this wave's columns, kept in VGPRs.
  v16h wf[2][8];
  #pragma unroll
  for (int nt = 0; nt < 2; ++nt)
    #pragma unroll
    for (int kc = 0; kc < 8; ++kc)
      wf[nt][kc] = load_wfrag(Wxh0, nbase + nt*16, kc, lane);
  const float bv0 = b0[nbase + (lane & 15)];
  const float bv1 = b0[nbase + 16 + (lane & 15)];

  __syncthreads();

  const int mo = (lane >> 4) * 8;   // C/D layout: lanes>=16 hold rows M+8
  const int nc = lane & 15;
  #pragma unroll
  for (int mt = 0; mt < 4; ++mt) {
    v8f c0 = zero8(), c1 = zero8();
    #pragma unroll
    for (int kc = 0; kc < 8; ++kc) {
      v16h a = load_afrag(&xt[mt * 16 * DH], kc, lane);
      c0 = wmma16(a, wf[0][kc], c0);
      c1 = wmma16(a, wf[1][kc], c1);
    }
    const size_t rbase = (size_t)t * BATCH + (size_t)bb * 64 + mt * 16;
    #pragma unroll
    for (int j = 0; j < 8; ++j) {
      const size_t r = rbase + mo + j;
      U0[r * DH + nbase + nc]      = (_Float16)(c0[j] + bv0);
      U0[r * DH + nbase + 16 + nc] = (_Float16)(c1[j] + bv1);
    }
  }
}

// ---------------------------------------------------------------------------
// Phase B: fused 2-layer scan + head.
// Grid: B/16 = 8 WGs (one WGP each); block 256 threads (8 waves, 2/SIMD).
// Per step (2 barriers):
//   issue async stage of U0[t+1]                (ASYNCcnt DMA, ping-pong buf)
//   c   = h0_prev @ Whh0^T                      (16 WMMA/wave)
//   h1  = tanh(c + U0[t]); store to LDS
//   c2  = h2_prev @ Whh1^T                      (independent of h1)
//   --- barrier (h1 visible) ---
//   c2 += h1 @ Wxh1^T;  h2 = tanh(c2 + b1); store to LDS
//   s_wait_asynccnt 0
//   --- barrier (step handoff) ---
// ---------------------------------------------------------------------------
__global__ __launch_bounds__(256, 1)
void rnn_scan_kernel(const _Float16* __restrict__ U0,
                     const float* __restrict__ Whh0, const float* __restrict__ Wxh1,
                     const float* __restrict__ Whh1, const float* __restrict__ b1,
                     const float* __restrict__ fcw,  const float* __restrict__ fcb,
                     float* __restrict__ out) {
  const int tid = threadIdx.x, lane = tid & 31, wave = tid >> 5;
  const int nbase  = wave * 32;     // this wave's 32 hidden columns
  const int bchunk = blockIdx.x;    // 16-row batch chunk

  __shared__ __attribute__((aligned(32))) _Float16 H0[2][16 * DH];   // layer-0 hidden (ping-pong)
  __shared__ __attribute__((aligned(32))) _Float16 H2[2][16 * DH];   // layer-1 hidden (ping-pong)
  __shared__ __attribute__((aligned(32))) _Float16 UB[2][16 * DH];   // staged U0 tiles (ping-pong)

  // Pin all scan weights in VGPRs (f32 -> f16 fragments, loaded once).
  v16h whh0[2][8], wxh1[2][8], whh1[2][8];
  #pragma unroll
  for (int nt = 0; nt < 2; ++nt)
    #pragma unroll
    for (int kc = 0; kc < 8; ++kc) {
      whh0[nt][kc] = load_wfrag(Whh0, nbase + nt*16, kc, lane);
      wxh1[nt][kc] = load_wfrag(Wxh1, nbase + nt*16, kc, lane);
      whh1[nt][kc] = load_wfrag(Whh1, nbase + nt*16, kc, lane);
    }
  const float b1v0 = b1[nbase + (lane & 15)];
  const float b1v1 = b1[nbase + 16 + (lane & 15)];

  const _Float16* uchunk = U0 + (size_t)bchunk * 16 * DH;   // +t*B*H per step

  for (int i = tid; i < 16 * DH; i += 256) {
    H0[0][i] = (_Float16)0.f;
    H2[0][i] = (_Float16)0.f;
  }
  // Pre-stage U0[0] into UB[0] (2 x 16B per thread = 8 KB, contiguous).
  async_copy16(&UB[0][tid * 8],        uchunk + tid * 8);
  async_copy16(&UB[0][tid * 8 + 2048], uchunk + tid * 8 + 2048);
  wait_async0();
  __syncthreads();

  const int mo = (lane >> 4) * 8;
  const int nc = lane & 15;

  for (int t = 0; t < T_STEPS; ++t) {
    const int p = t & 1;

    // Kick off DMA of the next step's U0 tile into the other buffer.
    if (t + 1 < T_STEPS) {
      const _Float16* nsrc = uchunk + (size_t)(t + 1) * (BATCH * DH);
      async_copy16(&UB[1 - p][tid * 8],        nsrc + tid * 8);
      async_copy16(&UB[1 - p][tid * 8 + 2048], nsrc + tid * 8 + 2048);
    }

    // ---- Layer 0 matmul: h0_prev @ Whh0^T
    v8f c0 = zero8(), c1 = zero8();
    #pragma unroll
    for (int kc = 0; kc < 8; ++kc) {
      v16h a = load_afrag(H0[p], kc, lane);
      c0 = wmma16(a, whh0[0][kc], c0);
      c1 = wmma16(a, whh0[1][kc], c1);
    }

    // h1 = tanh(c + U0[t]) -> LDS (other waves read it only after the barrier)
    _Float16* h1buf = H0[1 - p];
    #pragma unroll
    for (int j = 0; j < 8; ++j) {
      const int m = mo + j;
      float v0 = c0[j] + (float)UB[p][m * DH + nbase + nc];
      float v1 = c1[j] + (float)UB[p][m * DH + nbase + 16 + nc];
      h1buf[m * DH + nbase + nc]      = (_Float16)fast_tanh(v0);
      h1buf[m * DH + nbase + 16 + nc] = (_Float16)fast_tanh(v1);
    }

    // ---- Layer 1, recurrent half first (independent of h1_t)
    v8f d0 = zero8(), d1 = zero8();
    #pragma unroll
    for (int kc = 0; kc < 8; ++kc) {
      v16h a = load_afrag(H2[p], kc, lane);
      d0 = wmma16(a, whh1[0][kc], d0);
      d1 = wmma16(a, whh1[1][kc], d1);
    }

    __syncthreads();   // h1_t visible to all waves

    // ---- Layer 1, input half: + h1_t @ Wxh1^T
    #pragma unroll
    for (int kc = 0; kc < 8; ++kc) {
      v16h a = load_afrag(h1buf, kc, lane);
      d0 = wmma16(a, wxh1[0][kc], d0);
      d1 = wmma16(a, wxh1[1][kc], d1);
    }
    _Float16* h2buf = H2[1 - p];
    #pragma unroll
    for (int j = 0; j < 8; ++j) {
      const int m = mo + j;
      h2buf[m * DH + nbase + nc]      = (_Float16)fast_tanh(d0[j] + b1v0);
      h2buf[m * DH + nbase + 16 + nc] = (_Float16)fast_tanh(d1[j] + b1v1);
    }

    wait_async0();     // next U0 tile landed in LDS
    __syncthreads();   // step handoff
  }

  // Head: out[b,:] = h2[T-1] @ fcw^T + fcb for this chunk's 16 rows (OUT=2).
  // Final h2 lives in H2[0] (last write was t=1023, p=1 -> H2[0]).
  if (tid < 32) {
    const int r = tid & 15, o = tid >> 4;
    float s = fcb[o];
    const _Float16* h = &H2[0][r * DH];
    for (int k = 0; k < DH; ++k) s += (float)h[k] * fcw[o * DH + k];
    out[(size_t)(bchunk * 16 + r) * 2 + o] = s;
  }
}

// ---------------------------------------------------------------------------
extern "C" void kernel_launch(void* const* d_in, const int* in_sizes, int n_in,
                              void* d_out, int out_size, void* d_ws, size_t ws_size,
                              hipStream_t stream) {
  (void)in_sizes; (void)n_in; (void)out_size; (void)ws_size;
  const float* x    = (const float*)d_in[0];
  const float* Wxh0 = (const float*)d_in[1];
  const float* Whh0 = (const float*)d_in[2];
  const float* b0   = (const float*)d_in[3];
  const float* Wxh1 = (const float*)d_in[4];
  const float* Whh1 = (const float*)d_in[5];
  const float* b1   = (const float*)d_in[6];
  const float* fcw  = (const float*)d_in[7];
  const float* fcb  = (const float*)d_in[8];

  _Float16* U0 = (_Float16*)d_ws;   // T*B*H f16 = 32 MB scratch

  u0_gemm_kernel<<<dim3(T_STEPS, 2), 256, 0, stream>>>(x, Wxh0, b0, U0);
  rnn_scan_kernel<<<dim3(BATCH / 16), 256, 0, stream>>>(U0, Whh0, Wxh1, Whh1, b1,
                                                        fcw, fcb, (float*)d_out);
}
